// MultiHeadedAttention_64149631533354
// MI455X (gfx1250) — matrix operand order repaired and checked by reading, not compiled, not run on hardware
//
#include <hip/hip_runtime.h>

// MI455X (gfx1250) bf16-WMMA multi-head attention pipeline.
// query[2,2048,1024] @ W_in[1024,3072] -> qkv ; causal flash attention ;
// attn @ W_out[1024,1024] -> out (f32).
// Round 3: fix async builtin pointer types -> int4 with AS1 (global) / AS3 (LDS).

#define S_   2048
#define DM   1024
#define H_   16
#define HD   64
#define NROW 4096          // B*S
#define QKV  3072

typedef unsigned short ushort_t;
typedef unsigned int   uint_t;
typedef __attribute__((ext_vector_type(16))) __bf16 v16bf;
typedef __attribute__((ext_vector_type(8)))  float  v8f;
typedef __attribute__((ext_vector_type(4)))  int    v4i;

union FragU { v16bf v; uint4 q[2]; };

__device__ __forceinline__ ushort_t f2bf(float f) {
  union { float f; uint_t u; } c; c.f = f;
  uint_t u = c.u;
  return (ushort_t)((u + 0x7FFFu + ((u >> 16) & 1u)) >> 16);
}

// ---- CDNA5 async global->LDS staging (ASYNCcnt path) ----------------------
#if defined(__has_builtin)
#if __has_builtin(__builtin_amdgcn_global_load_async_to_lds_b128)
#define HAVE_ASYNC_LDS 1
#endif
#endif

#ifdef HAVE_ASYNC_LDS
typedef __attribute__((address_space(1))) v4i gv4i;   // global (AS1)
typedef __attribute__((address_space(3))) v4i lv4i;   // LDS (AS3)
__device__ __forceinline__ void stage16(const ushort_t* g, ushort_t* l) {
  // 16B per lane, memory -> LDS without touching VGPRs; tracked by ASYNCcnt.
  __builtin_amdgcn_global_load_async_to_lds_b128(
      (gv4i*)g, (lv4i*)l, 0, 0);
}
__device__ __forceinline__ void async_join() {
#if __has_builtin(__builtin_amdgcn_s_wait_asynccnt)
  __builtin_amdgcn_s_wait_asynccnt(0);
#else
  asm volatile("s_wait_asynccnt 0x0" ::: "memory");
#endif
}
#else
__device__ __forceinline__ void stage16(const ushort_t* g, ushort_t* l) {
  *(uint4*)l = *(const uint4*)g;
}
__device__ __forceinline__ void async_join() {}
#endif

__global__ void cast_bf16_kernel(const float* __restrict__ src,
                                 ushort_t* __restrict__ dst, int n) {
  int i = blockIdx.x * blockDim.x + threadIdx.x;
  if (i < n) dst[i] = f2bf(src[i]);
}

// ---------------------------------------------------------------------------
// GEMM1: X[4096,1024]bf16 x W_in[1024,3072]bf16 -> Q/K/V [b,h,s,64] bf16.
// Block = 128 threads (4 waves, 2x2), block tile 128x128, BK=32.
// LDS layouts are K-contiguous per row (A: m-major, B: n-major/transposed) so
// every WMMA fragment is exactly two b128 LDS loads per lane.
// ---------------------------------------------------------------------------
#define BM 128
#define BN 128
#define BK 32
#define LDT 40   // ushort stride (32 + 8 pad) -> 80B rows, 16B-aligned

__global__ __launch_bounds__(128) void gemm_qkv_kernel(
    const ushort_t* __restrict__ A,   // [4096][1024]
    const ushort_t* __restrict__ Bw,  // [1024][3072]
    ushort_t* __restrict__ Qo, ushort_t* __restrict__ Ko,
    ushort_t* __restrict__ Vo)
{
  __shared__ ushort_t As[BM * LDT];
  __shared__ ushort_t Bs[BN * LDT];
  const int t  = threadIdx.x;
  const int lane = t & 31, w = t >> 5;
  const int wm = w >> 1, wn = w & 1;
  const int m0 = blockIdx.x * BM;
  const int n0 = blockIdx.y * BN;
  const int lh = lane >> 4, ln = lane & 15;

  v8f acc[4][4];
  v8f zero = {};
#pragma unroll
  for (int i = 0; i < 4; i++)
#pragma unroll
    for (int j = 0; j < 4; j++) acc[i][j] = zero;

  for (int kt = 0; kt < DM; kt += BK) {
    __syncthreads();
    // Stage A: 128 rows x 32 k; pure copy -> async ASYNCcnt path.
#pragma unroll
    for (int it = 0; it < 4; it++) {
      int i = t + it * 128;                // 0..511
      int row = i >> 2, kq = i & 3;
      stage16(A + (size_t)(m0 + row) * DM + kt + kq * 8,
              As + row * LDT + kq * 8);
    }
    // Stage B transposed: Bs[n][k] (needs VGPR transpose at store).
#pragma unroll
    for (int it = 0; it < 4; it++) {
      int i = t + it * 128;
      int kr = i >> 4, nq = (i & 15) * 8;
      uint4 d = *(const uint4*)(Bw + (size_t)(kt + kr) * QKV + n0 + nq);
      const ushort_t* ds = (const ushort_t*)&d;
#pragma unroll
      for (int e = 0; e < 8; e++) Bs[(nq + e) * LDT + kr] = ds[e];
    }
    async_join();
    __syncthreads();

    FragU af[4], bf[4];
#pragma unroll
    for (int mt = 0; mt < 4; mt++) {
      // A frag: lanes 0-15 hold K {0..7,16..23}, lanes 16-31 {8..15,24..31}
      const ushort_t* p = As + (wm * 64 + mt * 16 + ln) * LDT + lh * 8;
      af[mt].q[0] = *(const uint4*)p;
      af[mt].q[1] = *(const uint4*)(p + 16);
    }
#pragma unroll
    for (int nt = 0; nt < 4; nt++) {
      // B frag: lanes 0-15 hold K {0..7,16..23}, lanes 16-31 {8..15,24..31}
      const ushort_t* p = Bs + (wn * 64 + nt * 16 + ln) * LDT + lh * 8;
      bf[nt].q[0] = *(const uint4*)p;
      bf[nt].q[1] = *(const uint4*)(p + 16);
    }
#pragma unroll
    for (int mt = 0; mt < 4; mt++)
#pragma unroll
      for (int nt = 0; nt < 4; nt++)
        acc[mt][nt] = __builtin_amdgcn_wmma_f32_16x16x32_bf16(
            false, af[mt].v, false, bf[nt].v, (short)0, acc[mt][nt],
            false, false);
  }

  // Epilogue: scatter to head-major Q/K/V; fold 1/sqrt(64) into Q.
#pragma unroll
  for (int nt = 0; nt < 4; nt++) {
    int col = n0 + wn * 64 + nt * 16 + ln;
    int which = col >> 10;               // 0=q 1=k 2=v
    int rem = col & 1023;
    int h = rem >> 6, dd = rem & 63;
    ushort_t* outp = (which == 0) ? Qo : ((which == 1) ? Ko : Vo);
    float sc = (which == 0) ? 0.125f : 1.0f;
#pragma unroll
    for (int mt = 0; mt < 4; mt++) {
#pragma unroll
      for (int r = 0; r < 8; r++) {
        int row = m0 + wm * 64 + mt * 16 + r + lh * 8;   // C layout
        int b = row >> 11, s = row & 2047;
        outp[(((size_t)(b * H_ + h) * S_ + s) << 6) + dd] =
            f2bf(acc[mt][nt][r] * sc);
      }
    }
  }
}

// ---------------------------------------------------------------------------
// Flash attention per (b,h): 64 q-rows per block (16 per wave), stream 64-key
// blocks through LDS with online softmax. Scores and P@V both on WMMA bf16.
// ---------------------------------------------------------------------------
#define KLD 72   // 64 + 8 pad halfs; 144B rows, 16B-aligned

__global__ __launch_bounds__(128) void attn_kernel(
    const ushort_t* __restrict__ Q, const ushort_t* __restrict__ K,
    const ushort_t* __restrict__ V, ushort_t* __restrict__ O)
{
  __shared__ ushort_t Ks[64 * KLD];          // [key][d]
  __shared__ ushort_t Vt[64 * KLD];          // [d][key] (transposed)
  __shared__ ushort_t Ps[4 * 16 * KLD];      // per-wave P scratch [row][key]
  const int t = threadIdx.x, lane = t & 31, w = t >> 5;
  const int lh = lane >> 4, ln = lane & 15;
  const int bh = blockIdx.y;                 // b*16+h
  const int qb0 = blockIdx.x * 64;
  const int qw = qb0 + w * 16;               // this wave's first q row
  const size_t base = (size_t)bh * S_ * HD;

  // Q fragments (A-operand, two K-chunks of 32). Q already pre-scaled.
  FragU qf[2];
#pragma unroll
  for (int kc = 0; kc < 2; kc++) {
    const ushort_t* p = Q + base + (size_t)(qw + ln) * HD + kc * 32 + lh * 8;
    qf[kc].q[0] = *(const uint4*)p;
    qf[kc].q[1] = *(const uint4*)(p + 16);
  }

  v8f o[4];
  v8f zero = {};
  float m[8], l[8];
#pragma unroll
  for (int i = 0; i < 4; i++) o[i] = zero;
#pragma unroll
  for (int i = 0; i < 8; i++) { m[i] = -1e30f; l[i] = 0.f; }

  ushort_t* Pw = Ps + w * 16 * KLD;
  const int jmax = qb0 >> 6;

  for (int j = 0; j <= jmax; j++) {
    const int jb = j * 64;
    __syncthreads();
    // Stage K (async copy, row-major) and V (transposed via VGPRs).
#pragma unroll
    for (int it = 0; it < 4; it++) {
      int i = t + it * 128;                  // 0..511
      int key = i >> 3, dq = (i & 7) * 8;
      stage16(K + base + (size_t)(jb + key) * HD + dq,
              Ks + key * KLD + dq);
      uint4 e = *(const uint4*)(V + base + (size_t)(jb + key) * HD + dq);
      const ushort_t* es = (const ushort_t*)&e;
#pragma unroll
      for (int x = 0; x < 8; x++) Vt[(dq + x) * KLD + key] = es[x];
    }
    async_join();
    __syncthreads();

    // Scores: S = Q(16x64) * K^T(64x64) as 4 sub-tiles of 16x16.
    v8f sacc[4];
#pragma unroll
    for (int nt = 0; nt < 4; nt++) {
      v8f s = zero;
#pragma unroll
      for (int kc = 0; kc < 2; kc++) {
        FragU kb;
        const ushort_t* p = Ks + (nt * 16 + ln) * KLD + kc * 32 + lh * 8;
        kb.q[0] = *(const uint4*)p;
        kb.q[1] = *(const uint4*)(p + 16);
        s = __builtin_amdgcn_wmma_f32_16x16x32_bf16(
            false, qf[kc].v, false, kb.v, (short)0, s, false, false);
      }
      sacc[nt] = s;
    }

    // Causal mask — only the diagonal key block needs it.
    if (jb + 63 > qw) {
      int rowb = qw + lh * 8;
#pragma unroll
      for (int nt = 0; nt < 4; nt++) {
        int col = jb + nt * 16 + ln;
#pragma unroll
        for (int r = 0; r < 8; r++)
          if (col > rowb + r) sacc[nt][r] = -1e30f;
      }
    }

    // Online softmax: row stats live per (VGPR r, lane-half) = one row each.
    float fac[8], rs[8];
#pragma unroll
    for (int r = 0; r < 8; r++) {
      float x = fmaxf(fmaxf(sacc[0][r], sacc[1][r]),
                      fmaxf(sacc[2][r], sacc[3][r]));
#pragma unroll
      for (int d_ = 1; d_ < 16; d_ <<= 1) x = fmaxf(x, __shfl_xor(x, d_, 32));
      float mn = fmaxf(m[r], x);
      fac[r] = __expf(m[r] - mn);
      m[r] = mn;
      rs[r] = 0.f;
    }
#pragma unroll
    for (int nt = 0; nt < 4; nt++) {
#pragma unroll
      for (int r = 0; r < 8; r++) {
        float p = __expf(sacc[nt][r] - m[r]);
        rs[r] += p;
        // C-layout -> LDS [row][key] for A-layout reload.
        Pw[(r + lh * 8) * KLD + nt * 16 + ln] = f2bf(p);
      }
    }
#pragma unroll
    for (int r = 0; r < 8; r++) {
#pragma unroll
      for (int d_ = 1; d_ < 16; d_ <<= 1) rs[r] += __shfl_xor(rs[r], d_, 32);
      l[r] = l[r] * fac[r] + rs[r];
    }

    // Rescale O and accumulate P(16x64) @ V(64x64) via WMMA (C-operand chain).
#pragma unroll
    for (int dt = 0; dt < 4; dt++)
#pragma unroll
      for (int r = 0; r < 8; r++) o[dt][r] *= fac[r];

    FragU pf[2];
#pragma unroll
    for (int kc = 0; kc < 2; kc++) {
      const ushort_t* p = Pw + ln * KLD + kc * 32 + lh * 8;
      pf[kc].q[0] = *(const uint4*)p;
      pf[kc].q[1] = *(const uint4*)(p + 16);
    }
#pragma unroll
    for (int dt = 0; dt < 4; dt++) {
#pragma unroll
      for (int kc = 0; kc < 2; kc++) {
        FragU vb;
        const ushort_t* p = Vt + (dt * 16 + ln) * KLD + kc * 32 + lh * 8;
        vb.q[0] = *(const uint4*)p;
        vb.q[1] = *(const uint4*)(p + 16);
        o[dt] = __builtin_amdgcn_wmma_f32_16x16x32_bf16(
            false, pf[kc].v, false, vb.v, (short)0, o[dt], false, false);
      }
    }
  }

  // Finalize: O /= l, write bf16 to merged [b,s,1024] layout.
  const int bq = bh >> 4, hq = bh & 15;
#pragma unroll
  for (int r = 0; r < 8; r++) {
    float inv = 1.0f / l[r];
    int srow = qw + r + lh * 8;
    ushort_t* op = O + ((size_t)(bq * S_ + srow)) * DM + hq * HD;
#pragma unroll
    for (int dt = 0; dt < 4; dt++)
      op[dt * 16 + ln] = f2bf(o[dt][r] * inv);
  }
}

// ---------------------------------------------------------------------------
// GEMM3: attn[4096,1024]bf16 x W_out[1024,1024]bf16 -> out f32.
// ---------------------------------------------------------------------------
__global__ __launch_bounds__(128) void gemm_out_kernel(
    const ushort_t* __restrict__ A,   // [4096][1024]
    const ushort_t* __restrict__ Bw,  // [1024][1024]
    float* __restrict__ C)
{
  __shared__ ushort_t As[BM * LDT];
  __shared__ ushort_t Bs[BN * LDT];
  const int t = threadIdx.x;
  const int lane = t & 31, w = t >> 5;
  const int wm = w >> 1, wn = w & 1;
  const int m0 = blockIdx.x * BM;
  const int n0 = blockIdx.y * BN;
  const int lh = lane >> 4, ln = lane & 15;

  v8f acc[4][4];
  v8f zero = {};
#pragma unroll
  for (int i = 0; i < 4; i++)
#pragma unroll
    for (int j = 0; j < 4; j++) acc[i][j] = zero;

  for (int kt = 0; kt < DM; kt += BK) {
    __syncthreads();
#pragma unroll
    for (int it = 0; it < 4; it++) {
      int i = t + it * 128;
      int row = i >> 2, kq = i & 3;
      stage16(A + (size_t)(m0 + row) * DM + kt + kq * 8,
              As + row * LDT + kq * 8);
    }
#pragma unroll
    for (int it = 0; it < 4; it++) {
      int i = t + it * 128;
      int kr = i >> 4, nq = (i & 15) * 8;
      uint4 d = *(const uint4*)(Bw + (size_t)(kt + kr) * DM + n0 + nq);
      const ushort_t* ds = (const ushort_t*)&d;
#pragma unroll
      for (int e = 0; e < 8; e++) Bs[(nq + e) * LDT + kr] = ds[e];
    }
    async_join();
    __syncthreads();

    FragU af[4], bf[4];
#pragma unroll
    for (int mt = 0; mt < 4; mt++) {
      const ushort_t* p = As + (wm * 64 + mt * 16 + ln) * LDT + lh * 8;
      af[mt].q[0] = *(const uint4*)p;
      af[mt].q[1] = *(const uint4*)(p + 16);
    }
#pragma unroll
    for (int nt = 0; nt < 4; nt++) {
      const ushort_t* p = Bs + (wn * 64 + nt * 16 + ln) * LDT + lh * 8;
      bf[nt].q[0] = *(const uint4*)p;
      bf[nt].q[1] = *(const uint4*)(p + 16);
    }
#pragma unroll
    for (int mt = 0; mt < 4; mt++)
#pragma unroll
      for (int nt = 0; nt < 4; nt++)
        acc[mt][nt] = __builtin_amdgcn_wmma_f32_16x16x32_bf16(
            false, af[mt].v, false, bf[nt].v, (short)0, acc[mt][nt],
            false, false);
  }

#pragma unroll
  for (int mt = 0; mt < 4; mt++) {
#pragma unroll
    for (int nt = 0; nt < 4; nt++) {
      int col = n0 + wn * 64 + nt * 16 + ln;
#pragma unroll
      for (int r = 0; r < 8; r++) {
        int row = m0 + wm * 64 + mt * 16 + r + lh * 8;
        C[(size_t)row * DM + col] = acc[mt][nt][r];
      }
    }
  }
}

// ---------------------------------------------------------------------------
extern "C" void kernel_launch(void* const* d_in, const int* in_sizes, int n_in,
                              void* d_out, int out_size, void* d_ws,
                              size_t ws_size, hipStream_t stream) {
  const float* query = (const float*)d_in[0];   // [2,2048,1024]
  const float* W_in  = (const float*)d_in[1];   // [1024,3072]
  const float* W_out = (const float*)d_in[2];   // [1024,1024]
  float* out = (float*)d_out;

  char* ws = (char*)d_ws;
  ushort_t* Xb = (ushort_t*)(ws);                    // 8 MiB
  ushort_t* Wi = (ushort_t*)(ws + (8u  << 20));      // 6 MiB
  ushort_t* Wo = (ushort_t*)(ws + (14u << 20));      // 2 MiB
  ushort_t* Qw = (ushort_t*)(ws + (16u << 20));      // 8 MiB
  ushort_t* Kw = (ushort_t*)(ws + (24u << 20));      // 8 MiB
  ushort_t* Vw = (ushort_t*)(ws + (32u << 20));      // 8 MiB
  ushort_t* Aw = (ushort_t*)(ws + (40u << 20));      // 8 MiB

  cast_bf16_kernel<<<(NROW * DM + 255) / 256, 256, 0, stream>>>(query, Xb,
                                                                NROW * DM);
  cast_bf16_kernel<<<(DM * QKV + 255) / 256, 256, 0, stream>>>(W_in, Wi,
                                                               DM * QKV);
  cast_bf16_kernel<<<(DM * DM + 255) / 256, 256, 0, stream>>>(W_out, Wo,
                                                              DM * DM);
  gemm_qkv_kernel<<<dim3(NROW / BM, QKV / BN), 128, 0, stream>>>(Xb, Wi, Qw,
                                                                 Kw, Vw);
  attn_kernel<<<dim3(S_ / 64, 2 * H_), 128, 0, stream>>>(Qw, Kw, Vw, Aw);
  gemm_out_kernel<<<dim3(NROW / BM, DM / BN), 128, 0, stream>>>(Aw, Wo, out);
}
